// DualWarping_18777597018141
// MI455X (gfx1250) — compile-verified, matrix operand order checked by reading, not run on hardware
//
#include <hip/hip_runtime.h>
#include <hip/hip_bf16.h>

typedef __attribute__((ext_vector_type(16))) __bf16 v16bf;
typedef __attribute__((ext_vector_type(8)))  float  v8f;
typedef __attribute__((ext_vector_type(4)))  unsigned int u32x4;
typedef __attribute__((ext_vector_type(4)))  float  f32x4;

#define HI 448
#define WI 768
#define HF 112
#define WF 192
#define HFP (HF + 2)
#define WFP (WF + 2)
#define NB 2
#define CF 96

__device__ __forceinline__ float clampf(float v, float lo, float hi) {
    return fminf(fmaxf(v, lo), hi);
}

// ---------------------------------------------------------------------------
// Kernel 0: zero-fill the padded NHWC activation zone (borders + wid ch3)
// ---------------------------------------------------------------------------
__global__ __launch_bounds__(256) void zerofill_kernel(u32x4* __restrict__ p, int n16)
{
    int idx = blockIdx.x * 256 + threadIdx.x;
    if (idx < n16) p[idx] = (u32x4){0u, 0u, 0u, 0u};
}

// ---------------------------------------------------------------------------
// Kernel 1: local_kernel_warp -> warped_img (B,3,H,W)
// ---------------------------------------------------------------------------
__global__ __launch_bounds__(256) void warp_img_kernel(
    const float* __restrict__ img, const float* __restrict__ flow,
    const float* __restrict__ kern, float* __restrict__ out)
{
    const int HW = HI * WI;
    int idx = blockIdx.x * 256 + threadIdx.x;
    if (idx >= NB * HW) return;
    int b = idx / HW;
    int pix = idx - b * HW;
    int y = pix / WI;
    int x = pix - y * WI;

    float vx = (float)x + flow[(size_t)(b * 2 + 0) * HW + pix];
    float vy = (float)y + flow[(size_t)(b * 2 + 1) * HW + pix];

    const float* p0 = img + (size_t)(b * 3 + 0) * HW;
    const float* p1 = img + (size_t)(b * 3 + 1) * HW;
    const float* p2 = img + (size_t)(b * 3 + 2) * HW;

    float a0 = 0.f, a1 = 0.f, a2 = 0.f;
    for (int t = 0; t < 25; ++t) {
        float relx = (float)(t % 5) - 2.0f;
        float rely = (float)(t / 5) - 2.0f;
        float xs = clampf(vx + relx, 0.f, (float)(WI - 1));
        float ys = clampf(vy + rely, 0.f, (float)(HI - 1));
        float x0f = floorf(xs), y0f = floorf(ys);
        float wx = xs - x0f, wy = ys - y0f;
        int ix0 = (int)x0f, iy0 = (int)y0f;
        int ix1 = min(ix0 + 1, WI - 1), iy1 = min(iy0 + 1, HI - 1);
        int i00 = iy0 * WI + ix0, i01 = iy0 * WI + ix1;
        int i10 = iy1 * WI + ix0, i11 = iy1 * WI + ix1;
        float kw = kern[(size_t)(b * 25 + t) * HW + pix];

        {   float v00 = p0[i00], v01 = p0[i01], v10 = p0[i10], v11 = p0[i11];
            float tp = v00 + (v01 - v00) * wx, bt = v10 + (v11 - v10) * wx;
            a0 += kw * (tp + (bt - tp) * wy); }
        {   float v00 = p1[i00], v01 = p1[i01], v10 = p1[i10], v11 = p1[i11];
            float tp = v00 + (v01 - v00) * wx, bt = v10 + (v11 - v10) * wx;
            a1 += kw * (tp + (bt - tp) * wy); }
        {   float v00 = p2[i00], v01 = p2[i01], v10 = p2[i10], v11 = p2[i11];
            float tp = v00 + (v01 - v00) * wx, bt = v10 + (v11 - v10) * wx;
            a2 += kw * (tp + (bt - tp) * wy); }
    }
    out[(size_t)(b * 3 + 0) * HW + pix] = a0;
    out[(size_t)(b * 3 + 1) * HW + pix] = a1;
    out[(size_t)(b * 3 + 2) * HW + pix] = a2;
}

// ---------------------------------------------------------------------------
// Kernel 2: backward_warp(features) -> wfeat f32 (NCHW) + act1p interior (NHWC96)
// ---------------------------------------------------------------------------
__global__ __launch_bounds__(256) void warp_feat_kernel(
    const float* __restrict__ feat, const float* __restrict__ flow,
    float* __restrict__ wfeat, __bf16* __restrict__ act1p)
{
    const int FHW = HF * WF;
    const int HW  = HI * WI;
    int idx = blockIdx.x * 256 + threadIdx.x;
    if (idx >= NB * FHW) return;
    int b = idx / FHW;
    int pix = idx - b * FHW;
    int yf = pix / WF;
    int xf = pix - yf * WF;

    float cy = clampf((float)yf * 4.0f + 1.5f, 0.f, (float)(HI - 1));
    float cx = clampf((float)xf * 4.0f + 1.5f, 0.f, (float)(WI - 1));
    int y0 = (int)floorf(cy); int y1 = min(y0 + 1, HI - 1); float wy = cy - (float)y0;
    int x0 = (int)floorf(cx); int x1 = min(x0 + 1, WI - 1); float wx = cx - (float)x0;

    const float* fx = flow + (size_t)(b * 2 + 0) * HW;
    const float* fy = flow + (size_t)(b * 2 + 1) * HW;
    float fsx, fsy;
    {
        float r0 = fx[y0 * WI + x0] * (1.f - wx) + fx[y0 * WI + x1] * wx;
        float r1 = fx[y1 * WI + x0] * (1.f - wx) + fx[y1 * WI + x1] * wx;
        fsx = (r0 * (1.f - wy) + r1 * wy) * 0.25f;
        r0 = fy[y0 * WI + x0] * (1.f - wx) + fy[y0 * WI + x1] * wx;
        r1 = fy[y1 * WI + x0] * (1.f - wx) + fy[y1 * WI + x1] * wx;
        fsy = (r0 * (1.f - wy) + r1 * wy) * 0.25f;
    }

    float vx = (float)xf + fsx;
    float vy = (float)yf + fsy;
    float valid = (vx >= 0.f && vx <= (float)(WF - 1) &&
                   vy >= 0.f && vy <= (float)(HF - 1)) ? 1.f : 0.f;
    float xs = clampf(vx, 0.f, (float)(WF - 1));
    float ys = clampf(vy, 0.f, (float)(HF - 1));
    float x0f = floorf(xs), y0f = floorf(ys);
    float swx = xs - x0f, swy = ys - y0f;
    int sx0 = (int)x0f, sy0 = (int)y0f;
    int sx1 = min(sx0 + 1, WF - 1), sy1 = min(sy0 + 1, HF - 1);
    int i00 = sy0 * WF + sx0, i01 = sy0 * WF + sx1;
    int i10 = sy1 * WF + sx0, i11 = sy1 * WF + sx1;

    size_t nhwc = ((size_t)(b * HFP + yf + 1) * WFP + xf + 1) * 96;
    for (int c = 0; c < CF; ++c) {
        const float* pc = feat + (size_t)(b * CF + c) * FHW;
        float v00 = pc[i00], v01 = pc[i01], v10 = pc[i10], v11 = pc[i11];
        float tp = v00 + (v01 - v00) * swx;
        float bt = v10 + (v11 - v10) * swx;
        float val = (tp + (bt - tp) * swy) * valid;
        wfeat[(size_t)(b * CF + c) * FHW + pix] = val;
        act1p[nhwc + c] = (__bf16)val;
    }
}

// ---------------------------------------------------------------------------
// Kernel 3: wid = resize_bilinear(warped_img) -> widp interior (NHWC4, ch3=0)
// ---------------------------------------------------------------------------
__global__ __launch_bounds__(256) void wid_kernel(
    const float* __restrict__ wimg, __bf16* __restrict__ widp)
{
    const int FHW = HF * WF;
    const int HW  = HI * WI;
    int idx = blockIdx.x * 256 + threadIdx.x;
    if (idx >= NB * FHW) return;
    int b = idx / FHW;
    int pix = idx - b * FHW;
    int yf = pix / WF;
    int xf = pix - yf * WF;

    float cy = clampf((float)yf * 4.0f + 1.5f, 0.f, (float)(HI - 1));
    float cx = clampf((float)xf * 4.0f + 1.5f, 0.f, (float)(WI - 1));
    int y0 = (int)floorf(cy); int y1 = min(y0 + 1, HI - 1); float wy = cy - (float)y0;
    int x0 = (int)floorf(cx); int x1 = min(x0 + 1, WI - 1); float wx = cx - (float)x0;
    int i00 = y0 * WI + x0, i01 = y0 * WI + x1;
    int i10 = y1 * WI + x0, i11 = y1 * WI + x1;

    size_t nhwc = ((size_t)(b * HFP + yf + 1) * WFP + xf + 1) * 4;
    for (int c = 0; c < 3; ++c) {
        const float* pc = wimg + (size_t)(b * 3 + c) * HW;
        float r0 = pc[i00] * (1.f - wx) + pc[i01] * wx;
        float r1 = pc[i10] * (1.f - wx) + pc[i11] * wx;
        widp[nhwc + c] = (__bf16)(r0 * (1.f - wy) + r1 * wy);
    }
}

// ---------------------------------------------------------------------------
// Kernel 4a: pack w1 (96,99,3,3) -> Bt1[n][896]: k<864: tap*96+cin(feat);
//            864+c*9+tap (wid chans); 891..895 zero.
// ---------------------------------------------------------------------------
__global__ __launch_bounds__(256) void pack_w1_kernel(
    const float* __restrict__ w, __bf16* __restrict__ dst)
{
    int idx = blockIdx.x * 256 + threadIdx.x;
    if (idx >= 96 * 896) return;
    int n = idx / 896;
    int k = idx - n * 896;
    float v = 0.f;
    if (k < 864) {
        int tap = k / 96, cin = k - tap * 96;
        v = w[(size_t)(n * 99 + cin) * 9 + tap];
    } else if (k < 891) {
        int kk = k - 864;
        int c = kk / 9, tap = kk - c * 9;
        v = w[(size_t)(n * 99 + 96 + c) * 9 + tap];
    }
    dst[idx] = (__bf16)v;
}

// Kernel 4b: pack w2 (96,96,3,3) -> Bt2[n][864]: k = tap*96+cin
__global__ __launch_bounds__(256) void pack_w2_kernel(
    const float* __restrict__ w, __bf16* __restrict__ dst)
{
    int idx = blockIdx.x * 256 + threadIdx.x;
    if (idx >= 96 * 864) return;
    int n = idx / 864;
    int k = idx - n * 864;
    int tap = k / 96, cin = k - tap * 96;
    dst[idx] = (__bf16)w[(size_t)(n * 96 + cin) * 9 + tap];
}

// ---------------------------------------------------------------------------
// Kernel 5: WMMA GEMM conv. Padded-NHWC activations -> unconditional async
// global->LDS B128 staging (fully unrolled issue bursts), double-buffered LDS,
// 6 back-to-back WMMAs per 32-wide K chunk.
//   CHMAJOR=1 (conv1): D = W x A, lane owns 8 contiguous channels -> NHWC bf16
//   CHMAJOR=0 (conv2): D = A x W, lane owns 8 contiguous pixels -> NCHW f32 + resid
// ---------------------------------------------------------------------------
template <int NC, bool HASWID, bool CHMAJOR>
__global__ __launch_bounds__(128) void conv3x3_wmma(
    const __bf16* __restrict__ act,   // (B,HFP,WFP,96) padded NHWC
    const __bf16* __restrict__ wid,   // (B,HFP,WFP,4) padded NHWC (conv1 only)
    const __bf16* __restrict__ wmat,  // Bt[96][NC*32]
    const float* __restrict__ bias,
    const float* __restrict__ resid,
    float* __restrict__ outf, __bf16* __restrict__ outbf)
{
    constexpr int KPAD = NC * 32;
    constexpr int NCF  = HASWID ? NC - 1 : NC;
    const int FHW = HF * WF;

    // rows padded to 40 bf16 = 80 B (16B-aligned, conflict-free fragments)
    __shared__ __align__(16) __bf16 AsBuf[2][64][40];
    __shared__ __align__(16) __bf16 BsBuf[2][96][40];

    const int tid  = threadIdx.x;
    const int wave = tid >> 5;
    const int lane = tid & 31;
    const int half = lane >> 4;
    const int lm   = lane & 15;

    const int mBase = blockIdx.x * 64;     // 64 | WF: tile = one image row segment
    const int bq = mBase / FHW;
    const int r2 = mBase - bq * FHW;
    const int yb = r2 / WF;
    const int xb = r2 - yb * WF;

    auto stage = [&](int nc, int buf) {
        // ---- B tile: 96 rows x 64 B, 3 async issues/thread (exact) ----
        {
            const int n0 = tid >> 2, part = tid & 3;
            const char* src0 = (const char*)wmat
                + ((size_t)n0 * KPAD + nc * 32) * 2 + part * 16;
            unsigned lds0 = (unsigned)(uintptr_t)((char*)&BsBuf[buf][n0][0] + part * 16);
#pragma unroll
            for (int it = 0; it < 3; ++it) {
                const char* src = src0 + (size_t)it * 32 * KPAD * 2;
                unsigned lds = lds0 + it * 32 * 80;
                asm volatile("global_load_async_to_lds_b128 %0, %1, off"
                             :: "v"(lds), "v"(src) : "memory");
            }
        }
        // ---- A tile ----
        if (!HASWID || nc < NCF) {
            // feat chunk: tap = nc/3, 32 contiguous channels per pixel
            int tap  = nc / 3;
            int cin0 = (nc - tap * 3) * 32;
            int ky = tap / 3, kx = tap - (tap / 3) * 3;
            size_t rowb = ((size_t)(bq * HFP + yb + ky) * WFP) * 96 + cin0;
            const int mi0 = tid >> 2, seg = tid & 3;
            const char* src0 = (const char*)act
                + (rowb + (size_t)(xb + mi0 + kx) * 96 + (size_t)seg * 8) * 2;
            unsigned lds0 = (unsigned)(uintptr_t)((char*)&AsBuf[buf][mi0][0] + seg * 16);
#pragma unroll
            for (int it = 0; it < 2; ++it) {
                const char* src = src0 + (size_t)it * 32 * 96 * 2;
                unsigned lds = lds0 + it * 32 * 80;
                asm volatile("global_load_async_to_lds_b128 %0, %1, off"
                             :: "v"(lds), "v"(src) : "memory");
            }
        } else {
            // wid chunk: kk = c*9 + tap (27 valid), batched unconditional gathers
            int kkw = tid & 31, mi0 = tid >> 5;
            if (kkw < 27) {
                int cc = kkw / 9, tap = kkw - cc * 9;
                int ky = tap / 3, kx = tap - (tap / 3) * 3;
                const __bf16* basep = wid
                    + ((size_t)(bq * HFP + yb + ky) * WFP + xb + kx) * 4 + cc;
                __bf16 vals[16];
#pragma unroll
                for (int m = 0; m < 16; ++m) vals[m] = basep[(size_t)(mi0 + 4 * m) * 4];
#pragma unroll
                for (int m = 0; m < 16; ++m) AsBuf[buf][mi0 + 4 * m][kkw] = vals[m];
            } else {
#pragma unroll
                for (int m = 0; m < 16; ++m) AsBuf[buf][mi0 + 4 * m][kkw] = (__bf16)0.f;
            }
        }
    };

    v8f acc[6];
#pragma unroll
    for (int j = 0; j < 6; ++j) acc[j] = (v8f){0.f, 0.f, 0.f, 0.f, 0.f, 0.f, 0.f, 0.f};

    stage(0, 0);
    asm volatile("s_wait_asynccnt 0" ::: "memory");
    __syncthreads();

    for (int i = 0; i < NC; ++i) {
        const int cur = i & 1;
        if (i + 1 < NC) stage(i + 1, cur ^ 1);   // prefetch while computing

        const __bf16* ap = &AsBuf[cur][wave * 16 + lm][half * 8];
        v16bf afrag;
#pragma unroll
        for (int e = 0; e < 8; ++e) { afrag[e] = ap[e]; afrag[e + 8] = ap[16 + e]; }
        v16bf wfr[6];
#pragma unroll
        for (int j = 0; j < 6; ++j) {
            const __bf16* bp = &BsBuf[cur][j * 16 + lm][half * 8];
#pragma unroll
            for (int e = 0; e < 8; ++e) { wfr[j][e] = bp[e]; wfr[j][e + 8] = bp[16 + e]; }
        }
#pragma unroll
        for (int j = 0; j < 6; ++j) {
            if (CHMAJOR)
                acc[j] = __builtin_amdgcn_wmma_f32_16x16x32_bf16(
                    false, wfr[j], false, afrag, (short)0, acc[j], false, false);
            else
                acc[j] = __builtin_amdgcn_wmma_f32_16x16x32_bf16(
                    false, afrag, false, wfr[j], (short)0, acc[j], false, false);
        }
        asm volatile("s_wait_asynccnt 0" ::: "memory");
        __syncthreads();
    }

    if (CHMAJOR) {
        // lane = one pixel; 8 contiguous channels per j -> one 16B NHWC store
        const int xx = xb + wave * 16 + lm;
        size_t pixb = ((size_t)(bq * HFP + yb + 1) * WFP + xx + 1) * 96;
#pragma unroll
        for (int j = 0; j < 6; ++j) {
            int ch0 = j * 16 + half * 8;
            f32x4 b0 = *(const f32x4*)(bias + ch0);
            f32x4 b1 = *(const f32x4*)(bias + ch0 + 4);
            float bv[8];
#pragma unroll
            for (int p = 0; p < 4; ++p) { bv[p] = b0[p]; bv[p + 4] = b1[p]; }
            unsigned pk[4];
#pragma unroll
            for (int p = 0; p < 4; ++p) {
                float v0 = fmaxf(acc[j][2 * p]     + bv[2 * p],     0.f);
                float v1 = fmaxf(acc[j][2 * p + 1] + bv[2 * p + 1], 0.f);
                unsigned short h0 = __builtin_bit_cast(unsigned short, (__bf16)v0);
                unsigned short h1 = __builtin_bit_cast(unsigned short, (__bf16)v1);
                pk[p] = (unsigned)h0 | ((unsigned)h1 << 16);
            }
            *(u32x4*)(outbf + pixb + ch0) = (u32x4){pk[0], pk[1], pk[2], pk[3]};
        }
    } else {
        // lane = one channel; 8 contiguous pixels per j -> float4 NCHW + resid
        const int xx0 = xb + wave * 16 + half * 8;
#pragma unroll
        for (int j = 0; j < 6; ++j) {
            int ch = j * 16 + lm;
            float bv = bias[ch];
            size_t base = ((size_t)(bq * 96 + ch) * HF + yb) * WF + xx0;
            f32x4 r0 = *(const f32x4*)(resid + base);
            f32x4 r1 = *(const f32x4*)(resid + base + 4);
            f32x4 o0, o1;
#pragma unroll
            for (int p = 0; p < 4; ++p) {
                o0[p] = r0[p] + acc[j][p] + bv;
                o1[p] = r1[p] + acc[j][p + 4] + bv;
            }
            *(f32x4*)(outf + base) = o0;
            *(f32x4*)(outf + base + 4) = o1;
        }
    }
}

// ---------------------------------------------------------------------------
extern "C" void kernel_launch(void* const* d_in, const int* in_sizes, int n_in,
                              void* d_out, int out_size, void* d_ws, size_t ws_size,
                              hipStream_t stream) {
    (void)in_sizes; (void)n_in; (void)out_size; (void)ws_size;
    const float* img  = (const float*)d_in[0];
    const float* feat = (const float*)d_in[1];
    const float* flow = (const float*)d_in[2];
    const float* kern = (const float*)d_in[3];
    const float* w1   = (const float*)d_in[4];
    const float* b1   = (const float*)d_in[5];
    const float* w2   = (const float*)d_in[6];
    const float* b2   = (const float*)d_in[7];

    const size_t FHW = (size_t)HF * WF;
    const size_t HW  = (size_t)HI * WI;
    const size_t PADHW = (size_t)HFP * WFP;          // 114*194
    float* out0 = (float*)d_out;                     // (B,96,HF,WF)
    float* out1 = out0 + (size_t)NB * CF * FHW;      // (B,3,HI,WI)

    char* ws = (char*)d_ws;
    size_t off = 0;
    auto carve = [&](size_t bytes) {
        char* p = ws + off;
        off = (off + bytes + 255) & ~(size_t)255;
        return p;
    };
    float*  wfeat = (float*)carve((size_t)NB * CF * FHW * sizeof(float));
    // padded-NHWC zone (zero-filled contiguously): act1p | act2p | widp
    const size_t zA1 = (size_t)NB * PADHW * 96 * sizeof(__bf16);   // mult of 256
    const size_t zA2 = (size_t)NB * PADHW * 96 * sizeof(__bf16);
    const size_t zW  = (size_t)NB * PADHW * 4  * sizeof(__bf16);
    __bf16* act1p = (__bf16*)carve(zA1);
    __bf16* act2p = (__bf16*)carve(zA2);
    __bf16* widp  = (__bf16*)carve(zW);
    __bf16* Bt1   = (__bf16*)carve((size_t)96 * 896 * sizeof(__bf16));
    __bf16* Bt2   = (__bf16*)carve((size_t)96 * 864 * sizeof(__bf16));

    const int n16 = (int)((zA1 + zA2 + zW) / 16);
    zerofill_kernel<<<(n16 + 255) / 256, 256, 0, stream>>>((u32x4*)act1p, n16);

    pack_w1_kernel<<<(96 * 896 + 255) / 256, 256, 0, stream>>>(w1, Bt1);
    pack_w2_kernel<<<(96 * 864 + 255) / 256, 256, 0, stream>>>(w2, Bt2);

    warp_img_kernel<<<(int)(NB * HW / 256), 256, 0, stream>>>(img, flow, kern, out1);
    warp_feat_kernel<<<(int)(NB * FHW / 256), 256, 0, stream>>>(feat, flow, wfeat, act1p);
    wid_kernel<<<(int)(NB * FHW / 256), 256, 0, stream>>>(out1, widp);

    const int gemmBlocks = (int)(NB * FHW / 64);   // 672
    conv3x3_wmma<28, true,  true ><<<gemmBlocks, 128, 0, stream>>>(
        act1p, widp, Bt1, b1, nullptr, nullptr, act2p);
    conv3x3_wmma<27, false, false><<<gemmBlocks, 128, 0, stream>>>(
        act2p, nullptr, Bt2, b2, wfeat, out0, nullptr);
}
